// Attention_29764123361713
// MI455X (gfx1250) — matrix operand order repaired and checked by reading, not compiled, not run on hardware
//
#include <hip/hip_runtime.h>

typedef __bf16 bf16;
typedef __attribute__((ext_vector_type(16))) __bf16 v16bf;
typedef __attribute__((ext_vector_type(8)))  __bf16 v8bf;
typedef __attribute__((ext_vector_type(2)))  __bf16 v2bf;
typedef __attribute__((ext_vector_type(8)))  float  v8f;
typedef __attribute__((ext_vector_type(8)))  unsigned v8u;

#define N_CTX 2048
#define HD    64
#define QTILE 128
#define KBLK  64
#define NBLK  (N_CTX / KBLK)

// ---- lane^16 exchange (cross-row) via v_permlanex16_b32, identity selects ----
#if __has_builtin(__builtin_amdgcn_permlanex16)
__device__ __forceinline__ unsigned xchg16u(unsigned x) {
  return (unsigned)__builtin_amdgcn_permlanex16(
      (int)x, (int)x, (int)0x76543210u, (int)0xFEDCBA98u, false, false);
}
#else
__device__ __forceinline__ unsigned xchg16u(unsigned x) {
  return (unsigned)__shfl_xor((int)x, 16, 32);
}
#endif
__device__ __forceinline__ float xchg16f(float x) {
  return __int_as_float((int)xchg16u((unsigned)__float_as_int(x)));
}

// ---- xor butterfly within 16-lane row (fallback kernel only) ----
#if __has_builtin(__builtin_amdgcn_permlane16)
#define PLX(x, s0, s1)                                                         \
  __int_as_float(__builtin_amdgcn_permlane16(                                  \
      __float_as_int(x), __float_as_int(x), (int)(s0), (int)(s1), false, false))
__device__ __forceinline__ float rowmax16(float v) {
  v = fmaxf(v, PLX(v, 0x67452301u, 0xEFCDAB89u));
  v = fmaxf(v, PLX(v, 0x54761032u, 0xDCFE98BAu));
  v = fmaxf(v, PLX(v, 0x32107654u, 0xBA98FEDCu));
  v = fmaxf(v, PLX(v, 0xFEDCBA98u, 0x76543210u));
  return v;
}
__device__ __forceinline__ float rowsum16(float v) {
  v += PLX(v, 0x67452301u, 0xEFCDAB89u);
  v += PLX(v, 0x54761032u, 0xDCFE98BAu);
  v += PLX(v, 0x32107654u, 0xBA98FEDCu);
  v += PLX(v, 0xFEDCBA98u, 0x76543210u);
  return v;
}
#else
__device__ __forceinline__ float rowmax16(float v) {
  for (int off = 8; off > 0; off >>= 1) v = fmaxf(v, __shfl_xor(v, off, 16));
  return v;
}
__device__ __forceinline__ float rowsum16(float v) {
  for (int off = 8; off > 0; off >>= 1) v += __shfl_xor(v, off, 16);
  return v;
}
#endif

__device__ __forceinline__ v16bf cat16(v8bf lo, v8bf hi) {
  return __builtin_shufflevector(lo, hi,
           0, 1, 2, 3, 4, 5, 6, 7, 8, 9, 10, 11, 12, 13, 14, 15);
}

// =====================================================================
// Pre-pass 1: plain f32 -> bf16 conversion (K, layout preserved)
// =====================================================================
__global__ __launch_bounds__(256) void cvt_bf16_kernel(
    const float* __restrict__ in, bf16* __restrict__ out) {
  const size_t i = ((size_t)blockIdx.x * 256 + threadIdx.x) * 8;
  v8f x = *(const v8f*)(in + i);
  *(v8bf*)(out + i) = __builtin_convertvector(x, v8bf);
}

// =====================================================================
// Pre-pass 2: V [head][n][d] f32 -> Vt [head][d][n] bf16 (transpose)
// =====================================================================
__global__ __launch_bounds__(256) void transpose_v_kernel(
    const float* __restrict__ V, bf16* __restrict__ Vt) {
  const int head = blockIdx.y;
  const int n = blockIdx.x * 256 + threadIdx.x;
  const float* src = V + ((size_t)head * N_CTX + n) * HD;
  bf16* dst = Vt + (size_t)head * HD * N_CTX + n;
#pragma unroll 8
  for (int d = 0; d < HD; ++d)
    dst[(size_t)d * N_CTX] = (bf16)src[d];
}

// =====================================================================
// Main kernel: transposed flash attention.
//   S^T = K * Q^T   (A = K tile,  B = Q^T, C = S^T: lane owns one q-row)
//   O^T = V^T * P^T (A = V^T tile, B = P^T assembled in registers)
// bf16 WMMA, async-to-LDS double-buffered staging, scalar softmax stats.
// =====================================================================
__global__ __launch_bounds__(256) void fa_fwd_async(
    const float* __restrict__ Q,
    const bf16* __restrict__ Kb,   // [head][n][d] bf16
    const bf16* __restrict__ Vt,   // [head][d][n] bf16
    float* __restrict__ O)
{
  __shared__ alignas(32) bf16 sK [2][KBLK][HD];     // 2 x 8 KB
  __shared__ alignas(32) bf16 sVt[2][HD][KBLK];     // 2 x 8 KB

  const int tid  = threadIdx.x;
  const int wave = tid >> 5;
  const int lane = tid & 31;
  const int m    = lane & 15;          // q-row within wave tile (both halves)
  const int half = lane >> 4;
  const int kb   = half << 3;          // A-layout K base: 0 or 8
  const bool hi  = half != 0;

  const size_t headOff = (size_t)blockIdx.y * (size_t)(N_CTX * HD);
  const int qrow0 = blockIdx.x * QTILE + wave * 16;

  // ---- Q^T in B-matrix 32x16 layout: lane = q-row, d = kb+0..7, kb+16..23 ----
  v16bf qreg[2];
  {
    const float* qp = Q + headOff + (size_t)(qrow0 + m) * HD;
    const float qscale = 0.125f * 1.44269504088896340736f;  // 1/sqrt(d)*log2e
#pragma unroll
    for (int ch = 0; ch < 2; ++ch) {
      v8f lo = *(const v8f*)(qp + ch * 32 + kb) * qscale;
      v8f hh = *(const v8f*)(qp + ch * 32 + kb + 16) * qscale;
      qreg[ch] = cat16(__builtin_convertvector(lo, v8bf),
                       __builtin_convertvector(hh, v8bf));
    }
  }

  v8f acc[4] = {};          // O^T d-tiles; lane col = q-row m
  float mrun = -1e30f, lrun = 0.0f;

  // ---- async staging setup ----
  const unsigned sKbase = (unsigned)(size_t)&sK[0][0][0];
  const unsigned sVbase = (unsigned)(size_t)&sVt[0][0][0];
  const char* Kg = (const char*)(Kb + headOff);
  const char* Vg = (const char*)(Vt + headOff);

  auto issue_async = [&](int blk, int buf) {
    if (wave < 4) {          // K block: 8 KB contiguous
      const char* g = Kg + (size_t)blk * (KBLK * HD * 2);
      const unsigned l0 = sKbase + (unsigned)(buf * (KBLK * HD * 2));
#pragma unroll
      for (int j = 0; j < 4; ++j) {
        const unsigned off = (unsigned)((wave * 4 + j) * 512 + lane * 16);
        const char* gp = g + off;
        const unsigned lp = l0 + off;
        asm volatile("global_load_async_to_lds_b128 %0, %1, off"
                     :: "v"(lp), "v"(gp) : "memory");
      }
    } else {                 // Vt block: 64 rows x 128 B, row stride N_CTX*2
      const int colb = (lane & 7) * 16;
      const char* g = Vg + (size_t)blk * (KBLK * 2) + colb;
      const unsigned l0 = sVbase + (unsigned)(buf * (KBLK * HD * 2) + colb);
#pragma unroll
      for (int j = 0; j < 4; ++j) {
        const int row = (wave - 4) * 16 + j * 4 + (lane >> 3);
        const char* gp = g + (size_t)row * (N_CTX * 2);
        const unsigned lp = l0 + (unsigned)(row * (KBLK * 2));
        asm volatile("global_load_async_to_lds_b128 %0, %1, off"
                     :: "v"(lp), "v"(gp) : "memory");
      }
    }
  };

  issue_async(0, 0);

  for (int blk = 0; blk < NBLK; ++blk) {
    const int buf = blk & 1;
    const int nb  = (blk + 1 < NBLK) ? blk + 1 : 0;
    issue_async(nb, buf ^ 1);

    asm volatile("s_wait_asynccnt 4" ::: "memory");
    __syncthreads();

    // ---- S^T = K * Q^T : 4 key-tiles; lane holds 8 keys per tile of row m ----
    v8f s[4] = {};
#pragma unroll
    for (int ch = 0; ch < 2; ++ch)
#pragma unroll
      for (int t = 0; t < 4; ++t) {
        const bf16* row = &sK[buf][t * 16 + m][0];
        const v16bf ak = cat16(*(const v8bf*)(row + ch * 32 + kb),
                               *(const v8bf*)(row + ch * 32 + kb + 16));
        s[t] = __builtin_amdgcn_wmma_f32_16x16x32_bf16(
                   false, ak, false, qreg[ch], (short)0, s[t], false, false);
      }

    // ---- scalar online softmax: one q-row per lane ----
    float vmax = s[0][0];
#pragma unroll
    for (int t = 0; t < 4; ++t)
#pragma unroll
      for (int r = 0; r < 8; ++r)
        vmax = fmaxf(vmax, s[t][r]);
    vmax = fmaxf(vmax, xchg16f(vmax));           // combine the two half-lanes

    const float mnew  = fmaxf(mrun, vmax);
    const float alpha = __builtin_amdgcn_exp2f(mrun - mnew);
    float rs = 0.0f;
#pragma unroll
    for (int t = 0; t < 4; ++t)
#pragma unroll
      for (int r = 0; r < 8; ++r) {
        const float p = __builtin_amdgcn_exp2f(s[t][r] - mnew);
        s[t][r] = p;
        rs += p;
      }
    rs += xchg16f(rs);
    lrun = lrun * alpha + rs;
    mrun = mnew;
#pragma unroll
    for (int dt = 0; dt < 4; ++dt)
      acc[dt] = acc[dt] * alpha;

    // ---- pack P to bf16 pairs: pk[t][v] = keys (t*16+kb+2v, +2v+1) of row m ----
    unsigned pk[4][4];
#pragma unroll
    for (int t = 0; t < 4; ++t)
#pragma unroll
      for (int v = 0; v < 4; ++v) {
        v2bf pr;
        pr.x = (bf16)s[t][2 * v];
        pr.y = (bf16)s[t][2 * v + 1];
        pk[t][v] = __builtin_bit_cast(unsigned, pr);
      }

    // ---- O^T += V^T * P^T : P^T B-operand assembled via permlanex16 ----
#pragma unroll
    for (int c = 0; c < 2; ++c) {
      // B 32x16: lanes 0-15 carry keys c*32+0..15 (tile 2c),
      //          lanes 16-31 carry keys c*32+16..31 (tile 2c+1)
      v8u bu;
#pragma unroll
      for (int v = 0; v < 4; ++v) {
        const unsigned lo  = pk[2 * c][v];
        const unsigned hh  = pk[2 * c + 1][v];
        const unsigned xlo = xchg16u(lo);
        const unsigned xhh = xchg16u(hh);
        bu[v]     = hi ? xhh : lo;    // keys +0..7  of this lane's tile
        bu[v + 4] = hi ? hh  : xlo;   // keys +8..15 of this lane's tile
      }
      const v16bf pb = __builtin_bit_cast(v16bf, bu);
#pragma unroll
      for (int dt = 0; dt < 4; ++dt) {
        const bf16* row = &sVt[buf][dt * 16 + m][0];
        const v16bf av = cat16(*(const v8bf*)(row + c * 32 + half * 16),
                               *(const v8bf*)(row + c * 32 + half * 16 + 8));
        acc[dt] = __builtin_amdgcn_wmma_f32_16x16x32_bf16(
                      false, av, false, pb, (short)0, acc[dt], false, false);
      }
    }
    __syncthreads();
  }

  // ---- normalize and write: 4 contiguous v8f stores per lane ----
  const float rinv = 1.0f / lrun;
  float* op = O + headOff + (size_t)(qrow0 + m) * HD;
#pragma unroll
  for (int dt = 0; dt < 4; ++dt) {
    v8f o = acc[dt] * rinv;
    *(v8f*)(op + dt * 16 + half * 8) = o;
  }
}

// =====================================================================
// Fallback (no workspace): fused kernel, in-kernel conversion
// =====================================================================
__global__ __launch_bounds__(256) void fa_fwd_fused(
    const float* __restrict__ Q,
    const float* __restrict__ K,
    const float* __restrict__ V,
    float* __restrict__ O)
{
  __shared__ alignas(32) bf16 sK [KBLK][HD];
  __shared__ alignas(32) bf16 sVt[HD][KBLK];
  __shared__ alignas(32) bf16 sP [8][16][KBLK];

  const int tid  = threadIdx.x;
  const int wave = tid >> 5;
  const int lane = tid & 31;
  const int m    = lane & 15;
  const int half = lane >> 4;
  const int kb   = half << 3;

  const size_t headOff = (size_t)blockIdx.y * (size_t)(N_CTX * HD);
  const int qrow0 = blockIdx.x * QTILE + wave * 16;

  v16bf qreg[2];
  {
    const float* qp = Q + headOff + (size_t)(qrow0 + m) * HD;
    const float qscale = 0.125f * 1.44269504088896340736f;
#pragma unroll
    for (int c = 0; c < 2; ++c) {
      v8f lo = *(const v8f*)(qp + c * 32 + half * 16) * qscale;
      v8f hi = *(const v8f*)(qp + c * 32 + half * 16 + 8) * qscale;
      qreg[c] = cat16(__builtin_convertvector(lo, v8bf),
                      __builtin_convertvector(hi, v8bf));
    }
  }

  v8f acc[4] = {};
  float mrun[8], lrun[8];
#pragma unroll
  for (int r = 0; r < 8; ++r) { mrun[r] = -1e30f; lrun[r] = 0.0f; }

  const int krow  = tid >> 2;
  const int kdcol = (tid & 3) << 4;
  const int vkey  = (tid >> 3) << 1;
  const int vd0   = (tid & 7) << 3;

  const float* Kbase = K + headOff;
  const float* Vbase = V + headOff;

  v8f pk0, pk1, pva, pvb;
  {
    const float* kp = Kbase + (size_t)krow * HD + kdcol;
    pk0 = *(const v8f*)kp;
    pk1 = *(const v8f*)(kp + 8);
    const float* vp = Vbase + (size_t)vkey * HD + vd0;
    pva = *(const v8f*)vp;
    pvb = *(const v8f*)(vp + HD);
  }

  for (int blk = 0; blk < NBLK; ++blk) {
    *(v16bf*)&sK[krow][kdcol] = cat16(__builtin_convertvector(pk0, v8bf),
                                      __builtin_convertvector(pk1, v8bf));
    v8bf a = __builtin_convertvector(pva, v8bf);
    v8bf b = __builtin_convertvector(pvb, v8bf);
#pragma unroll
    for (int i = 0; i < 8; ++i) {
      v2bf pr; pr.x = a[i]; pr.y = b[i];
      *(v2bf*)&sVt[vd0 + i][vkey] = pr;
    }
    __syncthreads();

    {
      const int nb = (blk + 1 < NBLK) ? blk + 1 : 0;
      const float* kp = Kbase + (size_t)(nb * KBLK + krow) * HD + kdcol;
      pk0 = *(const v8f*)kp;
      pk1 = *(const v8f*)(kp + 8);
      const float* vp = Vbase + (size_t)(nb * KBLK + vkey) * HD + vd0;
      pva = *(const v8f*)vp;
      pvb = *(const v8f*)(vp + HD);
    }

    v8f s[4] = {};
#pragma unroll
    for (int ch = 0; ch < 2; ++ch)
#pragma unroll
      for (int t = 0; t < 4; ++t) {
        const v16bf bk = *(const v16bf*)&sK[t * 16 + m][ch * 32 + half * 16];
        s[t] = __builtin_amdgcn_wmma_f32_16x16x32_bf16(
                   false, qreg[ch], false, bk, (short)0, s[t], false, false);
      }

#pragma unroll
    for (int r = 0; r < 8; ++r) {
      float v = fmaxf(fmaxf(s[0][r], s[1][r]), fmaxf(s[2][r], s[3][r]));
      v = rowmax16(v);
      const float mnew  = fmaxf(mrun[r], v);
      const float alpha = __builtin_amdgcn_exp2f(mrun[r] - mnew);
      float rs = 0.0f;
#pragma unroll
      for (int t = 0; t < 4; ++t) {
        const float p = __builtin_amdgcn_exp2f(s[t][r] - mnew);
        s[t][r] = p;
        rs += p;
      }
      rs = rowsum16(rs);
      lrun[r] = lrun[r] * alpha + rs;
      mrun[r] = mnew;
#pragma unroll
      for (int dt = 0; dt < 4; ++dt)
        acc[dt][r] *= alpha;
    }

#pragma unroll
    for (int t = 0; t < 4; ++t)
#pragma unroll
      for (int r = 0; r < 8; ++r)
        sP[wave][half * 8 + r][t * 16 + m] = (bf16)s[t][r];
    asm volatile("s_wait_dscnt 0" ::: "memory");

    v16bf pa[2];
#pragma unroll
    for (int c = 0; c < 2; ++c) {
      v8bf plo = *(const v8bf*)&sP[wave][m][c * 32 + half * 16];
      v8bf phi = *(const v8bf*)&sP[wave][m][c * 32 + half * 16 + 8];
      pa[c] = cat16(plo, phi);
    }

#pragma unroll
    for (int c = 0; c < 2; ++c)
#pragma unroll
      for (int dt = 0; dt < 4; ++dt) {
        const v16bf bv = *(const v16bf*)&sVt[dt * 16 + m][c * 32 + half * 16];
        acc[dt] = __builtin_amdgcn_wmma_f32_16x16x32_bf16(
                      false, pa[c], false, bv, (short)0, acc[dt], false, false);
      }
    __syncthreads();
  }

  float rinv[8];
#pragma unroll
  for (int r = 0; r < 8; ++r) rinv[r] = 1.0f / lrun[r];

  float* op = O + headOff;
#pragma unroll
  for (int dt = 0; dt < 4; ++dt)
#pragma unroll
    for (int r = 0; r < 8; ++r) {
      const int row = qrow0 + half * 8 + r;
      op[(size_t)row * HD + dt * 16 + m] = acc[dt][r] * rinv[r];
    }
}

extern "C" void kernel_launch(void* const* d_in, const int* in_sizes, int n_in,
                              void* d_out, int out_size, void* d_ws, size_t ws_size,
                              hipStream_t stream) {
  const float* Q = (const float*)d_in[0];
  const float* K = (const float*)d_in[1];
  const float* V = (const float*)d_in[2];
  float* O = (float*)d_out;

  const size_t elems = (size_t)in_sizes[1];          // B*H*N*d
  const int heads = (int)(elems / (N_CTX * HD));     // B*H = 32
  dim3 grid(N_CTX / QTILE, heads);
  dim3 block(256);

  const size_t need = 2 * elems * sizeof(bf16);      // Kbf + Vt
  if (ws_size >= need) {
    bf16* Kbf = (bf16*)d_ws;
    bf16* Vtb = Kbf + elems;
    cvt_bf16_kernel<<<dim3((unsigned)(elems / (256 * 8))), block, 0, stream>>>(K, Kbf);
    transpose_v_kernel<<<dim3(N_CTX / 256, heads), block, 0, stream>>>(V, Vtb);
    fa_fwd_async<<<grid, block, 0, stream>>>(Q, Kbf, Vtb, O);
  } else {
    fa_fwd_fused<<<grid, block, 0, stream>>>(Q, K, V, O);
  }
}
